// Encoder_62405874811323
// MI455X (gfx1250) — matrix operand order repaired and checked by reading, not compiled, not run on hardware
//
#include <hip/hip_runtime.h>
#include <hip/hip_bf16.h>

typedef __attribute__((ext_vector_type(16))) _Float16 v16h;
typedef __attribute__((ext_vector_type(8)))  _Float16 v8h;
typedef __attribute__((ext_vector_type(8)))  float    v8f;

constexpr int kH   = 128;
constexpr int kT   = 120;
constexpr int kFin = 180;
constexpr int kKP  = 192;   // FC K padded to multiple of 32
constexpr int kB   = 2048;
constexpr int RS0  = 200;   // FC weight LDS row stride (halves), bank-skewed
constexpr int RS1  = 264;   // LSTM weight LDS row stride (halves), bank-skewed
constexpr int RSH  = 136;   // activation LDS row stride (halves): 272B = 68 dwords, skew 4 banks/lane
constexpr int BT   = 32;    // batch rows per workgroup in LSTM pass

// LSTM-kernel dynamic LDS layout (byte offsets)
constexpr unsigned OFF_H    = 512u * RS1 * 2;              // hbuf after weights
constexpr unsigned OFF_BIAS = OFF_H + (unsigned)BT * RSH * 2;
constexpr unsigned OFF_X    = OFF_BIAS + 512u * 4;         // x_t double buffer
constexpr unsigned XBUF_SZ  = (unsigned)BT * RSH * 2;      // one x buffer
constexpr size_t   SMEM_B   = OFF_X + 2u * XBUF_SZ;        // = 298496 B

__device__ __forceinline__ v16h mk16(v8h lo, v8h hi) {
  return __builtin_shufflevector(lo, hi, 0,1,2,3,4,5,6,7,8,9,10,11,12,13,14,15);
}
__device__ __forceinline__ v8f wmma_f16(v16h a, v16h b, v8f c) {
  // D = A(16x32 f16) x B(32x16 f16) + C(16x16 f32)
  return __builtin_amdgcn_wmma_f32_16x16x32_f16(false, a, false, b, (short)0, c,
                                                false, false);
}
__device__ __forceinline__ float sigm(float x) { return 1.0f / (1.0f + __expf(-x)); }
__device__ __forceinline__ float tanh_fast(float x) {
  float e = __expf(-2.0f * x);
  return (1.0f - e) / (1.0f + e);
}

// ---------------------------------------------------------------------------
// Kernel 1: u[(b*T+t), h] = LeakyReLU(BN(FC(x[b, :, t])))  -> f16
// Grid: (B*T/128) blocks x 256 threads. Wave w: 16 rows, all 8 col tiles.
// ---------------------------------------------------------------------------
__global__ __launch_bounds__(256) void fc_bn_act_kernel(
    const float* __restrict__ x, const float* __restrict__ fc_w,
    const float* __restrict__ fc_b, const float* __restrict__ bn_g,
    const float* __restrict__ bn_b, const float* __restrict__ bn_mean,
    const float* __restrict__ bn_var, _Float16* __restrict__ u) {
  __shared__ _Float16 Wl[kH * RS0];
  __shared__ float alpha[kH];
  __shared__ float beta[kH];
  const int tid = threadIdx.x;

  for (int idx = tid; idx < kH * kKP; idx += 256) {
    int hc = idx / kKP, k = idx % kKP;
    float v = (k < kFin) ? fc_w[hc * kFin + k] : 0.0f;
    Wl[hc * RS0 + k] = (_Float16)v;
  }
  if (tid < kH) {
    float sc = bn_g[tid] * rsqrtf(bn_var[tid] + 1e-5f);
    alpha[tid] = sc;
    beta[tid] = (fc_b[tid] - bn_mean[tid]) * sc + bn_b[tid];
  }
  __syncthreads();

  const int w = tid >> 5, L = tid & 31, g = L >> 4, ln = L & 15;
  const int m0 = blockIdx.x * 128 + w * 16;
  const int mrow = m0 + ln;                 // A-fragment row for this lane
  const int b = mrow / kT, t = mrow - b * kT;
  const float* __restrict__ xrow = x + (size_t)b * kFin * kT + t;

  v8f acc[8];
#pragma unroll
  for (int nt = 0; nt < 8; ++nt)
#pragma unroll
    for (int r = 0; r < 8; ++r) acc[nt][r] = 0.0f;

#pragma unroll
  for (int ch = 0; ch < kKP / 32; ++ch) {
    v16h a;
#pragma unroll
    for (int e = 0; e < 8; ++e) {
      int k1 = 32 * ch + g * 8 + e;        // elems 0..7 ; k1 <= 175 always valid
      int k2 = k1 + 16;                    // elems 8..15; may exceed 179 (last chunk)
      float v1 = xrow[(size_t)k1 * kT];
      int k2c = (k2 < kFin) ? k2 : (kFin - 1);
      float v2 = xrow[(size_t)k2c * kT];
      v2 = (k2 < kFin) ? v2 : 0.0f;
      a[e] = (_Float16)v1;
      a[e + 8] = (_Float16)v2;
    }
#pragma unroll
    for (int nt = 0; nt < 8; ++nt) {
      const _Float16* bp = &Wl[(nt * 16 + ln) * RS0 + 32 * ch + g * 8];
      acc[nt] = wmma_f16(a, mk16(*(const v8h*)bp, *(const v8h*)(bp + 16)), acc[nt]);
    }
  }

#pragma unroll
  for (int nt = 0; nt < 8; ++nt) {
    int col = nt * 16 + ln;
    float al = alpha[col], be = beta[col];
#pragma unroll
    for (int r = 0; r < 8; ++r) {
      int m = m0 + r + 8 * g;
      float v = acc[nt][r] * al + be;
      v = (v >= 0.0f) ? v : 0.01f * v;
      u[(size_t)m * kH + col] = (_Float16)v;
    }
  }
}

// ---------------------------------------------------------------------------
// Kernel 2: one LSTM layer, persistent per batch tile of 32 rows.
// gates(32x512) = [x_t | h_prev](32x256) @ [wih;whh]^T, K=256, f16 WMMA.
// x_t tiles are double-buffered in LDS via async global->LDS copies
// (GLOBAL_LOAD_ASYNC_TO_LDS_B128, ASYNCcnt), prefetched one step ahead.
// ---------------------------------------------------------------------------
__global__ __launch_bounds__(256) void lstm_layer_kernel(
    const _Float16* __restrict__ in_seq, const float* __restrict__ wih,
    const float* __restrict__ whh, const float* __restrict__ bih,
    const float* __restrict__ bhh, _Float16* __restrict__ out_seq,
    float* __restrict__ out_last, int write_seq, int write_last) {
  extern __shared__ char smem[];
  _Float16* Wl   = (_Float16*)smem;                 // 512 x RS1 halves
  _Float16* hbuf = (_Float16*)(smem + OFF_H);       // 32 x RSH halves
  float*    bias = (float*)(smem + OFF_BIAS);       // 512 floats
  _Float16* xbuf = (_Float16*)(smem + OFF_X);       // 2 x 32 x RSH halves
  const int tid = threadIdx.x;
  const int w = tid >> 5, L = tid & 31, g = L >> 4, ln = L & 15;
  const int b0 = blockIdx.x * BT;
  const int seg = L & 15;                           // 16B segment within a row
  const unsigned long long inbase = (unsigned long long)(uintptr_t)in_seq;

  // Issue async copy of x_{tt} tile (32 rows x 256B) into LDS buffer `buf`.
  // 8 waves x 2 issues x 32 lanes x 16B = 8 KB.
  auto issue_xtile = [&](int tt, int buf) {
#pragma unroll
    for (int j = 0; j < 2; ++j) {
      int row = 4 * w + 2 * j + (L >> 4);
      unsigned go = (unsigned)(((b0 + row) * kT + tt) * (kH * 2) + seg * 16);
      unsigned lo = OFF_X + (unsigned)buf * XBUF_SZ + (unsigned)row * (RSH * 2) +
                    (unsigned)seg * 16;
      asm volatile("global_load_async_to_lds_b128 %0, %1, %2"
                   :: "v"(lo), "v"(go), "s"(inbase) : "memory");
    }
  };

  issue_xtile(0, 0);  // preload t = 0 while we fill weights

  for (int idx = tid; idx < 512 * 256; idx += 256) {
    int row = idx >> 8, k = idx & 255;
    float v = (k < 128) ? wih[row * 128 + k] : whh[row * 128 + (k - 128)];
    Wl[row * RS1 + k] = (_Float16)v;
  }
  for (int idx = tid; idx < 512; idx += 256) bias[idx] = bih[idx] + bhh[idx];
  for (int idx = tid; idx < BT * RSH; idx += 256) hbuf[idx] = (_Float16)0.0f;
  asm volatile("s_wait_asynccnt 0x0" ::: "memory");
  __syncthreads();

  const int col = w * 16 + ln;                 // H-column owned by this lane

  float bsg[4];
#pragma unroll
  for (int gi = 0; gi < 4; ++gi) bsg[gi] = bias[gi * 128 + col];

  v8f cst[2];                                  // cell-state tiles (mt = 0,1)
#pragma unroll
  for (int mt = 0; mt < 2; ++mt)
#pragma unroll
    for (int r = 0; r < 8; ++r) cst[mt][r] = 0.0f;

  for (int t = 0; t < kT; ++t) {
    const int buf = t & 1;
    if (t + 1 < kT) issue_xtile(t + 1, buf ^ 1);   // overlap with this step's WMMAs
    const _Float16* xb = xbuf + buf * (BT * RSH);

    v8f acc[4][2];
#pragma unroll
    for (int gi = 0; gi < 4; ++gi)
#pragma unroll
      for (int mt = 0; mt < 2; ++mt)
#pragma unroll
        for (int r = 0; r < 8; ++r) acc[gi][mt][r] = 0.0f;

#pragma unroll
    for (int ch = 0; ch < 8; ++ch) {
      const _Float16* abase = (ch < 4) ? xb : hbuf;
      const int koff = (ch & 3) * 32 + g * 8;
      v16h a[2];
#pragma unroll
      for (int mt = 0; mt < 2; ++mt) {
        const _Float16* p = abase + (mt * 16 + ln) * RSH + koff;
        a[mt] = mk16(*(const v8h*)p, *(const v8h*)(p + 16));
      }
#pragma unroll
      for (int gi = 0; gi < 4; ++gi) {
        const _Float16* bp = Wl + (size_t)(gi * 128 + col) * RS1 + ch * 32 + g * 8;
        v16h bf = mk16(*(const v8h*)bp, *(const v8h*)(bp + 16));
        acc[gi][0] = wmma_f16(a[0], bf, acc[gi][0]);
        acc[gi][1] = wmma_f16(a[1], bf, acc[gi][1]);
      }
    }
    __syncthreads();                           // all reads of hbuf complete

#pragma unroll
    for (int mt = 0; mt < 2; ++mt) {
#pragma unroll
      for (int r = 0; r < 8; ++r) {
        float iv = sigm(acc[0][mt][r] + bsg[0]);
        float fv = sigm(acc[1][mt][r] + bsg[1]);
        float gv = tanh_fast(acc[2][mt][r] + bsg[2]);
        float ov = sigm(acc[3][mt][r] + bsg[3]);
        float cv = fv * cst[mt][r] + iv * gv;
        cst[mt][r] = cv;
        float hv = ov * tanh_fast(cv);
        int m = mt * 16 + r + 8 * g;           // local batch row of D element
        hbuf[m * RSH + col] = (_Float16)hv;
        if (write_seq)
          out_seq[((size_t)(b0 + m) * kT + t) * kH + col] = (_Float16)hv;
        if (write_last && t == kT - 1)
          out_last[(size_t)(b0 + m) * kH + col] = hv;
      }
    }
    asm volatile("s_wait_asynccnt 0x0" ::: "memory");  // x_{t+1} tile landed
    __syncthreads();                           // hbuf + xbuf ready for next step
  }
}

// ---------------------------------------------------------------------------
extern "C" void kernel_launch(void* const* d_in, const int* in_sizes, int n_in,
                              void* d_out, int out_size, void* d_ws, size_t ws_size,
                              hipStream_t stream) {
  const float* x       = (const float*)d_in[0];
  const float* fc_w    = (const float*)d_in[1];
  const float* fc_b    = (const float*)d_in[2];
  const float* bn_g    = (const float*)d_in[3];
  const float* bn_b    = (const float*)d_in[4];
  const float* bn_mean = (const float*)d_in[5];
  const float* bn_var  = (const float*)d_in[6];
  const float* wih0    = (const float*)d_in[7];
  const float* whh0    = (const float*)d_in[8];
  const float* bih0    = (const float*)d_in[9];
  const float* bhh0    = (const float*)d_in[10];
  const float* wih1    = (const float*)d_in[11];
  const float* whh1    = (const float*)d_in[12];
  const float* bih1    = (const float*)d_in[13];
  const float* bhh1    = (const float*)d_in[14];

  _Float16* u  = (_Float16*)d_ws;                         // (B*T, H) f16
  _Float16* h0 = u + (size_t)kB * kT * kH;                // (B*T, H) f16

  (void)hipFuncSetAttribute((const void*)lstm_layer_kernel,
                            hipFuncAttributeMaxDynamicSharedMemorySize,
                            (int)SMEM_B);

  const int rows = kB * kT;                               // 245760
  fc_bn_act_kernel<<<rows / 128, 256, 0, stream>>>(
      x, fc_w, fc_b, bn_g, bn_b, bn_mean, bn_var, u);

  lstm_layer_kernel<<<kB / BT, 256, SMEM_B, stream>>>(
      u, wih0, whh0, bih0, bhh0, h0, nullptr, 1, 0);

  lstm_layer_kernel<<<kB / BT, 256, SMEM_B, stream>>>(
      h0, wih1, whh1, bih1, bhh1, nullptr, (float*)d_out, 0, 1);
}
